// GraphUNetTopK_44985487458609
// MI455X (gfx1250) — compile-verified, matrix operand order checked by reading, not run on hardware
//
#include <hip/hip_runtime.h>
#include <hip/hip_bf16.h>
#include <math.h>

// ---------------- problem constants (match reference) ----------------
#define GB     512
#define GN     128
#define GFEAT  128
#define GNCLS  10
#define GDEG   16
#define GE     (GB*GN*GDEG)      // 1,048,576 edges
#define PK1    103
#define PK2    83
#define PK3    67
#define M0V    (GB*GN)           // 65536
#define M1V    (GB*PK1)          // 52736 = 128*412
#define M2V    (GB*PK2)          // 42496 = 128*332
#define M3V    (GB*PK3)          // 34304
#define BNEPS  1e-5f
#define NEGINF (-3.402823466e38f)
#define KMAX   256               // largest GEMM K (dec3.l1)

typedef __attribute__((ext_vector_type(16))) __bf16 bf16x16;
typedef __attribute__((ext_vector_type(8)))  __bf16 bf16x8;
typedef __attribute__((ext_vector_type(8)))  float  f32x8;

__device__ __forceinline__ __bf16 f2bf(float f) {
  union { float f; unsigned u; } x; x.f = f;
  unsigned short h = (unsigned short)((x.u + 0x7FFFu + ((x.u >> 16) & 1u)) >> 16);
  return __builtin_bit_cast(__bf16, h);
}

// Async global->LDS copy, 16B per lane (CDNA5, tracked by ASYNCcnt).
__device__ __forceinline__ void async_b128(unsigned ldsOff, const float* gptr) {
  asm volatile("global_load_async_to_lds_b128 %0, %1, off"
               :: "v"(ldsOff), "v"(gptr) : "memory");
}

// ---------------- WMMA GEMM: C[MxN] = A[MxK] @ W[KxN] + bias ----------------
// Block = 256 thr (8 waves) owns a 128x16 output panel (M%128==0, N%16==0,
// K%32==0 by construction). Weights are converted to bf16 once per block into
// LDS (transposed). Each wave double-buffers its 16x32 f32 A-tile through LDS
// with GLOBAL_LOAD_ASYNC_TO_LDS_B128 + s_wait_asynccnt.
__global__ void __launch_bounds__(256)
wmma_gemm_bias_kernel(const float* __restrict__ A, const float* __restrict__ W,
                      const float* __restrict__ bias, float* __restrict__ C,
                      int M, int K, int N)
{
  __shared__ float  lds_a[8 * 2 * 512];      // 8 waves x 2 bufs x (16x32 f32) = 32KB
  __shared__ __bf16 lds_wt[16 * KMAX];       // W^T stripe: 16 cols x K bf16   = 8KB

  int tid    = (int)threadIdx.x;
  int wave   = tid >> 5;
  int lane   = tid & 31;
  int tilesN = N >> 4;
  int bm     = (int)blockIdx.x / tilesN;
  int bn     = ((int)blockIdx.x % tilesN) << 4;

  // ---- stage W^T stripe (cols bn..bn+15) as bf16 in LDS, once per block ----
  for (int idx = tid; idx < 16 * K; idx += 256) {
    int n = idx & 15;        // consecutive tid -> consecutive n (coalesced gl. read)
    int k = idx >> 4;
    lds_wt[n * K + k] = f2bf(W[(size_t)k * N + bn + n]);
  }
  __syncthreads();

  int row0 = (bm << 7) + (wave << 4);
  if (row0 >= M) return;                     // wave-uniform (never taken: M%128==0)

  int half = lane >> 4;
  int r    = lane & 15;
  int col  = bn + r;

  // ---- per-lane async source pointers: 4 x 16B chunks cover the 16x32 tile ----
  const float* gsrc[4];
  unsigned     ldsd[4];
  unsigned sliceOff = (unsigned)(size_t)&lds_a[wave * 1024];   // LDS byte offset
#pragma unroll
  for (int q = 0; q < 4; ++q) {
    int chunk = q * 32 + lane;               // 0..127 16B-chunks
    int arow  = chunk >> 3;                  // 0..15
    int ac4   = (chunk & 7) << 2;            // float offset within row
    gsrc[q] = A + (size_t)(row0 + arow) * K + ac4;
    ldsd[q] = sliceOff + (unsigned)chunk * 16u;
  }
  const float*  afp = &lds_a[wave * 1024 + r * 32 + half * 8]; // A-frag base (buf0)
  const __bf16* bfp = &lds_wt[r * K + half * 8];               // B-frag base

  // prologue: request tile k0=0 into buf0
#pragma unroll
  for (int q = 0; q < 4; ++q) async_b128(ldsd[q], gsrc[q]);

  f32x8 acc = {0.f,0.f,0.f,0.f,0.f,0.f,0.f,0.f};
  int buf = 0;
  for (int k0 = 0; k0 < K; k0 += 32) {
    if (k0 + 32 < K) {                       // request next tile into other buffer
#pragma unroll
      for (int q = 0; q < 4; ++q)
        async_b128(ldsd[q] + (unsigned)((buf ^ 1) * 2048), gsrc[q] + (k0 + 32));
      asm volatile("s_wait_asynccnt 0x4" ::: "memory");  // current tile resident
    } else {
      asm volatile("s_wait_asynccnt 0x0" ::: "memory");
    }

    // A fragment: 16 f32 from LDS (4 x ds_load_b128), convert to bf16
    const float* as = afp + buf * 512;
    float4 a0 = *(const float4*)(as);
    float4 a1 = *(const float4*)(as + 4);
    float4 a2 = *(const float4*)(as + 16);
    float4 a3 = *(const float4*)(as + 20);
    bf16x16 af;
    af[0]=f2bf(a0.x); af[1]=f2bf(a0.y); af[2]=f2bf(a0.z);  af[3]=f2bf(a0.w);
    af[4]=f2bf(a1.x); af[5]=f2bf(a1.y); af[6]=f2bf(a1.z);  af[7]=f2bf(a1.w);
    af[8]=f2bf(a2.x); af[9]=f2bf(a2.y); af[10]=f2bf(a2.z); af[11]=f2bf(a2.w);
    af[12]=f2bf(a3.x);af[13]=f2bf(a3.y);af[14]=f2bf(a3.z); af[15]=f2bf(a3.w);

    // B fragment: two contiguous 16B bf16 chunks from LDS
    bf16x8 b0 = *(const bf16x8*)(bfp + k0);
    bf16x8 b1 = *(const bf16x8*)(bfp + k0 + 16);
    bf16x16 bv;
#pragma unroll
    for (int j = 0; j < 8; ++j) { bv[j] = b0[j]; bv[j + 8] = b1[j]; }

    acc = __builtin_amdgcn_wmma_f32_16x16x32_bf16(
        false, af, false, bv, (short)0, acc, false, false);
    buf ^= 1;
  }

  float bvv = bias[col];
#pragma unroll
  for (int v = 0; v < 8; ++v) {
    int m = row0 + half * 8 + v;             // ISA C/D layout: VGPR v -> M = v+half*8
    C[(size_t)m * N + col] = acc[v] + bvv;
  }
}

// ---------------- BatchNorm pass 1: per-column sum / sumsq ----------------
__global__ void colstats_kernel(const float* __restrict__ H, float* __restrict__ stats,
                                int M, int N)
{
  __shared__ float ssum[256];
  __shared__ float ssq[256];
  int n = blockIdx.x;
  float s = 0.f, q = 0.f;
  for (int m = (int)threadIdx.x; m < M; m += 256) {
    float v = H[(size_t)m * N + n];
    s += v; q += v * v;
  }
  ssum[threadIdx.x] = s; ssq[threadIdx.x] = q;
  __syncthreads();
  for (int off = 128; off > 0; off >>= 1) {
    if ((int)threadIdx.x < off) {
      ssum[threadIdx.x] += ssum[threadIdx.x + off];
      ssq[threadIdx.x]  += ssq[threadIdx.x + off];
    }
    __syncthreads();
  }
  if (threadIdx.x == 0) { stats[n] = ssum[0]; stats[N + n] = ssq[0]; }
}

// ---------------- BatchNorm pass 2 + ReLU (in place) ----------------
__global__ void bn_relu_kernel(float* __restrict__ H, const float* __restrict__ stats,
                               const float* __restrict__ g, const float* __restrict__ be,
                               int M, int N)
{
  int i = blockIdx.x * blockDim.x + threadIdx.x;
  if (i >= M * N) return;
  int n = i % N;
  float fm  = (float)M;
  float mu  = stats[n] / fm;
  float var = stats[N + n] / fm - mu * mu;
  float v = g[n] * (H[i] - mu) * rsqrtf(var + BNEPS) + be[n];
  H[i] = v > 0.f ? v : 0.f;
}

// ---------------- helpers ----------------
__global__ void copy4_kernel(float4* __restrict__ d, const float4* __restrict__ s, int n4)
{
  int i = blockIdx.x * blockDim.x + threadIdx.x;
  if (i < n4) d[i] = s[i];
}

__global__ void add_kernel(float* __restrict__ d, const float* __restrict__ a,
                           const float* __restrict__ b, int n)
{
  int i = blockIdx.x * blockDim.x + threadIdx.x;
  if (i < n) d[i] = a[i] + b[i];
}

// ---------------- GIN edge aggregation: aggr[dst] += X[src] (masked) -----
__global__ void edge_aggr_kernel(float* __restrict__ aggr, const float* __restrict__ X,
                                 const int* __restrict__ src, const int* __restrict__ dst,
                                 const int* __restrict__ mask, int lgC)
{
  int i = blockIdx.x * blockDim.x + threadIdx.x;   // exactly GE<<lgC threads launched
  int e = i >> lgC;
  int c = i & ((1 << lgC) - 1);
  if (mask && !mask[e]) return;
  float v = X[((size_t)src[e] << lgC) + c];
  unsafeAtomicAdd(&aggr[((size_t)dst[e] << lgC) + c], v);
}

// ---------------- TopK pooling: one 128-thread block per graph ----------
__global__ void topk_pool_kernel(const float* __restrict__ X, const float* __restrict__ p,
                                 float* __restrict__ Xnew, int* __restrict__ inv,
                                 int n_cur, int k, int C)
{
  __shared__ float sp[128];
  __shared__ float red[128];
  __shared__ float sval[128];
  __shared__ int   sidx[128];
  int b = blockIdx.x, t = (int)threadIdx.x;

  float pv = (t < C) ? p[t] : 0.f;
  sp[t]  = pv;
  red[t] = pv * pv;
  __syncthreads();
  for (int off = 64; off > 0; off >>= 1) {
    if (t < off) red[t] += red[t + off];
    __syncthreads();
  }
  float invn = rsqrtf(red[0]);

  float s = NEGINF;
  if (t < n_cur) {
    const float* xr = X + (size_t)(b * n_cur + t) * C;
    float acc = 0.f;
    for (int c = 0; c < C; ++c) acc += xr[c] * sp[c];
    s = acc * invn;
  }
  sval[t] = s; sidx[t] = t;
  __syncthreads();

  // bitonic sort, descending
  for (int sz = 2; sz <= 128; sz <<= 1) {
    for (int st = sz >> 1; st > 0; st >>= 1) {
      int j = t ^ st;
      if (j > t) {
        bool desc = ((t & sz) == 0);
        float v0 = sval[t], v1 = sval[j];
        bool sw = desc ? (v0 < v1) : (v0 > v1);
        if (sw) {
          sval[t] = v1; sval[j] = v0;
          int tmp = sidx[t]; sidx[t] = sidx[j]; sidx[j] = tmp;
        }
      }
      __syncthreads();
    }
  }

  if (t < n_cur) inv[b * n_cur + t] = -1;
  __syncthreads();
  if (t < k) {
    int nid = sidx[t];
    float gate = tanhf(sval[t]);
    inv[b * n_cur + nid] = b * k + t;
    const float* xr = X + (size_t)(b * n_cur + nid) * C;
    float* yr = Xnew + (size_t)(b * k + t) * C;
    for (int c = 0; c < C; ++c) yr[c] = xr[c] * gate;
  }
}

// ---------------- edge relabel after pooling ----------------
__global__ void relabel_kernel(const int* __restrict__ src, const int* __restrict__ dst,
                               const int* __restrict__ mask,
                               int* __restrict__ nsrc, int* __restrict__ ndst,
                               int* __restrict__ nmask, const int* __restrict__ inv)
{
  int e = blockIdx.x * blockDim.x + threadIdx.x;
  if (e >= GE) return;
  int ok = mask ? mask[e] : 1;
  int ns = 0, nd = 0;
  if (ok) {
    ns = inv[src[e]];
    nd = inv[dst[e]];
    ok = (ns >= 0) && (nd >= 0);
  }
  nsrc[e]  = ok ? ns : 0;
  ndst[e]  = ok ? nd : 0;
  nmask[e] = ok;
}

// ---------------- readout: [max | mean] over k nodes per graph ----------
__global__ void readout_kernel(const float* __restrict__ X, float* __restrict__ out,
                               int k, int C)
{
  int i = blockIdx.x * blockDim.x + threadIdx.x;
  if (i >= GB * C) return;
  int b = i / C, c = i % C;
  const float* xp = X + (size_t)b * k * C + c;
  float mx = NEGINF, sm = 0.f;
  for (int j = 0; j < k; ++j) {
    float v = xp[(size_t)j * C];
    mx = fmaxf(mx, v); sm += v;
  }
  out[(size_t)b * 2 * C + c]     = mx;
  out[(size_t)b * 2 * C + C + c] = sm / (float)k;
}

// ---------------- final Linear(64->10) + log_softmax ----------------
__global__ void dec1_ls_kernel(const float* __restrict__ X, const float* __restrict__ W,
                               const float* __restrict__ b, float* __restrict__ out)
{
  int i = blockIdx.x * blockDim.x + threadIdx.x;
  if (i >= GB) return;
  const float* xr = X + (size_t)i * 64;
  float lg[GNCLS];
  float mx = NEGINF;
#pragma unroll
  for (int c = 0; c < GNCLS; ++c) {
    float s = b[c];
    for (int kk = 0; kk < 64; ++kk) s += xr[kk] * W[kk * GNCLS + c];
    lg[c] = s; mx = fmaxf(mx, s);
  }
  float den = 0.f;
#pragma unroll
  for (int c = 0; c < GNCLS; ++c) den += expf(lg[c] - mx);
  float l = logf(den);
#pragma unroll
  for (int c = 0; c < GNCLS; ++c) out[(size_t)i * GNCLS + c] = lg[c] - mx - l;
}

// =====================================================================
extern "C" void kernel_launch(void* const* d_in, const int* in_sizes, int n_in,
                              void* d_out, int out_size, void* d_ws, size_t ws_size,
                              hipStream_t stream)
{
  (void)in_sizes; (void)n_in; (void)out_size; (void)ws_size;

  // ---- inputs (setup_inputs dict order, leaves depth-first) ----
  const float* x0   = (const float*)d_in[0];
  const int*   ei   = (const int*)d_in[1];
  const int* src0 = ei;
  const int* dst0 = ei + GE;
  const float* c1l1W = (const float*)d_in[3];
  const float* c1l1b = (const float*)d_in[4];
  const float* c1g1  = (const float*)d_in[5];
  const float* c1be1 = (const float*)d_in[6];
  const float* c1l2W = (const float*)d_in[7];
  const float* c1l2b = (const float*)d_in[8];
  const float* c1g2  = (const float*)d_in[9];
  const float* c1be2 = (const float*)d_in[10];
  const float* p1    = (const float*)d_in[11];
  const float* c2l1W = (const float*)d_in[12];
  const float* c2l1b = (const float*)d_in[13];
  const float* c2g1  = (const float*)d_in[14];
  const float* c2be1 = (const float*)d_in[15];
  const float* c2l2W = (const float*)d_in[16];
  const float* c2l2b = (const float*)d_in[17];
  const float* c2g2  = (const float*)d_in[18];
  const float* c2be2 = (const float*)d_in[19];
  const float* p2    = (const float*)d_in[20];
  const float* c3l1W = (const float*)d_in[21];
  const float* c3l1b = (const float*)d_in[22];
  const float* c3g1  = (const float*)d_in[23];
  const float* c3be1 = (const float*)d_in[24];
  const float* c3l2W = (const float*)d_in[25];
  const float* c3l2b = (const float*)d_in[26];
  const float* c3g2  = (const float*)d_in[27];
  const float* c3be2 = (const float*)d_in[28];
  const float* p3    = (const float*)d_in[29];
  const float* d3l1W = (const float*)d_in[30];
  const float* d3l1b = (const float*)d_in[31];
  const float* d3g1  = (const float*)d_in[32];
  const float* d3be1 = (const float*)d_in[33];
  const float* d3l2W = (const float*)d_in[34];
  const float* d3l2b = (const float*)d_in[35];
  const float* d3g2  = (const float*)d_in[36];
  const float* d3be2 = (const float*)d_in[37];
  const float* d2l1W = (const float*)d_in[38];
  const float* d2l1b = (const float*)d_in[39];
  const float* d2g1  = (const float*)d_in[40];
  const float* d2be1 = (const float*)d_in[41];
  const float* d2l2W = (const float*)d_in[42];
  const float* d2l2b = (const float*)d_in[43];
  const float* d2g2  = (const float*)d_in[44];
  const float* d2be2 = (const float*)d_in[45];
  const float* d1W   = (const float*)d_in[46];
  const float* d1b   = (const float*)d_in[47];

  // ---- workspace carve-up (buffers reused across stages) ----
  char* wsp = (char*)d_ws;
  auto take = [&](size_t bytes) -> char* {
    char* pp = wsp; wsp += (bytes + 255) & ~(size_t)255; return pp;
  };
  float* fA   = (float*)take(sizeof(float) * (size_t)M0V * GFEAT);  // 32 MB
  float* fB   = (float*)take(sizeof(float) * (size_t)M3V * GFEAT);  // 17.6 MB
  float* fC   = (float*)take(sizeof(float) * (size_t)M2V * GFEAT);  // 21.8 MB
  float* stats= (float*)take(sizeof(float) * 256);
  float* R1   = (float*)take(sizeof(float) * GB * 64);
  float* R2   = (float*)take(sizeof(float) * GB * 128);
  float* R3   = (float*)take(sizeof(float) * GB * 256);
  float* Dx   = (float*)take(sizeof(float) * GB * 128);
  float* Dy   = (float*)take(sizeof(float) * GB * 128);
  float* Dz   = (float*)take(sizeof(float) * GB * 128);
  int*   iInv = (int*)take(sizeof(int) * M0V);
  int*   e1s  = (int*)take(sizeof(int) * GE);
  int*   e1d  = (int*)take(sizeof(int) * GE);
  int*   e1m  = (int*)take(sizeof(int) * GE);
  int*   e2s  = (int*)take(sizeof(int) * GE);
  int*   e2d  = (int*)take(sizeof(int) * GE);
  int*   e2m  = (int*)take(sizeof(int) * GE);

  auto cdiv = [](long long a, long long b) { return (unsigned)((a + b - 1) / b); };
  auto gemm = [&](const float* A, const float* W, const float* bias, float* C,
                  int M, int K, int N) {
    unsigned blocks = (unsigned)((M >> 7) * (N >> 4));   // 128-row x 16-col panels
    wmma_gemm_bias_kernel<<<blocks, 256, 0, stream>>>(A, W, bias, C, M, K, N);
  };
  auto bn = [&](float* H, const float* g, const float* be, int M, int N) {
    colstats_kernel<<<N, 256, 0, stream>>>(H, stats, M, N);
    bn_relu_kernel<<<cdiv((long long)M * N, 256), 256, 0, stream>>>(H, stats, g, be, M, N);
  };
  auto copyf = [&](float* d, const float* s, long long n) {
    copy4_kernel<<<cdiv(n / 4, 256), 256, 0, stream>>>((float4*)d, (const float4*)s, (int)(n / 4));
  };
  auto aggr = [&](float* A, const float* X, const int* s_, const int* d_,
                  const int* m_, int lgC) {
    edge_aggr_kernel<<<cdiv((long long)GE << lgC, 256), 256, 0, stream>>>(A, X, s_, d_, m_, lgC);
  };

  // ================= stage 1: GIN(128->32) + pool K1 =================
  copyf(fA, x0, (long long)M0V * GFEAT);                 // aggr := x
  aggr(fA, x0, src0, dst0, nullptr, 7);                  // += neighbor msgs (C=128)
  gemm(fA, c1l1W, c1l1b, fB, M0V, 128, 32);  bn(fB, c1g1, c1be1, M0V, 32);
  gemm(fB, c1l2W, c1l2b, fA, M0V, 32, 32);   bn(fA, c1g2, c1be2, M0V, 32);
  topk_pool_kernel<<<GB, 128, 0, stream>>>(fA, p1, fB, iInv, GN, PK1, 32);
  relabel_kernel<<<cdiv(GE, 256), 256, 0, stream>>>(src0, dst0, nullptr, e1s, e1d, e1m, iInv);
  readout_kernel<<<cdiv(GB * 32, 256), 256, 0, stream>>>(fB, R1, PK1, 32);

  // ================= stage 2: GIN(32->64) + pool K2 =================
  copyf(fA, fB, (long long)M1V * 32);
  aggr(fA, fB, e1s, e1d, e1m, 5);                        // C=32
  gemm(fA, c2l1W, c2l1b, fC, M1V, 32, 64);   bn(fC, c2g1, c2be1, M1V, 64);
  gemm(fC, c2l2W, c2l2b, fA, M1V, 64, 64);   bn(fA, c2g2, c2be2, M1V, 64);
  topk_pool_kernel<<<GB, 128, 0, stream>>>(fA, p2, fB, iInv, PK1, PK2, 64);
  relabel_kernel<<<cdiv(GE, 256), 256, 0, stream>>>(e1s, e1d, e1m, e2s, e2d, e2m, iInv);
  readout_kernel<<<cdiv(GB * 64, 256), 256, 0, stream>>>(fB, R2, PK2, 64);

  // ================= stage 3: GIN(64->128) + pool K3 =================
  copyf(fA, fB, (long long)M2V * 64);
  aggr(fA, fB, e2s, e2d, e2m, 6);                        // C=64
  gemm(fA, c3l1W, c3l1b, fC, M2V, 64, 128);  bn(fC, c3g1, c3be1, M2V, 128);
  gemm(fC, c3l2W, c3l2b, fA, M2V, 128, 128); bn(fA, c3g2, c3be2, M2V, 128);
  topk_pool_kernel<<<GB, 128, 0, stream>>>(fA, p3, fB, iInv, PK2, PK3, 128);
  // (relabeled edges after pool3 are never used by the reference output)
  readout_kernel<<<cdiv(GB * 128, 256), 256, 0, stream>>>(fB, R3, PK3, 128);

  // ================= decoder =================
  gemm(R3, d3l1W, d3l1b, Dx, GB, 256, 128);  bn(Dx, d3g1, d3be1, GB, 128);
  gemm(Dx, d3l2W, d3l2b, Dy, GB, 128, 128);  bn(Dy, d3g2, d3be2, GB, 128);
  add_kernel<<<cdiv(GB * 128, 256), 256, 0, stream>>>(Dx, Dy, R2, GB * 128);
  gemm(Dx, d2l1W, d2l1b, Dy, GB, 128, 64);   bn(Dy, d2g1, d2be1, GB, 64);
  gemm(Dy, d2l2W, d2l2b, Dz, GB, 64, 64);    bn(Dz, d2g2, d2be2, GB, 64);
  add_kernel<<<cdiv(GB * 64, 256), 256, 0, stream>>>(Dy, Dz, R1, GB * 64);
  dec1_ls_kernel<<<cdiv(GB, 256), 256, 0, stream>>>(Dy, d1W, d1b, (float*)d_out);
}
